// MatchesLayerDistillationSegmentorV4_84361747628522
// MI455X (gfx1250) — compile-verified, matrix operand order checked by reading, not run on hardware
//
#include <hip/hip_runtime.h>
#include <hip/hip_bf16.h>

typedef __attribute__((ext_vector_type(16))) _Float16 v16h;
typedef __attribute__((ext_vector_type(8)))  float    v8f;

#define N_PTS  16384
#define N_ENC  8192
#define NCLS   22
#define CB     64
#define S_DIM  256
#define T_DIM  512
#define BN_EPS 1e-3f

// ------------------------------------------------------------------
// helpers
// ------------------------------------------------------------------
// pack two f32 -> one dword of two f16 (v_cvt_pk_f16_f32)
__device__ __forceinline__ unsigned int pack2h(float a, float b) {
    union { _Float16 h[2]; unsigned int u; } r;
    r.h[0] = (_Float16)a;
    r.h[1] = (_Float16)b;
    return r.u;
}

// LDS tile layout: row-major, K packed as f16 pairs (1 dword = 2 halfs).
// Fragment for lane (hi = lane>>4) at k-step kk:
//   halfs 0..7  = K  kk+hi*8   .. kk+hi*8+7    -> 4 dwords at  base
//   halfs 8..15 = K  kk+16+hi*8.. kk+hi*8+23   -> 4 dwords at  base+8
// where base = row*(K/2) + kk/2 + hi*4   (two ds_load_b128)
__device__ __forceinline__ v16h frag_from_lds(const unsigned int* p) {
    union { unsigned int u[8]; v16h h; } r;
#pragma unroll
    for (int i = 0; i < 4; ++i) {
        r.u[i]     = p[i];
        r.u[4 + i] = p[8 + i];
    }
    return r.h;
}

// ------------------------------------------------------------------
// zero accumulator / BN-stat region
// ------------------------------------------------------------------
__global__ void zero_kernel(float* p, int n) {
    int i = blockIdx.x * blockDim.x + threadIdx.x;
    if (i < n) p[i] = 0.f;
}

// ------------------------------------------------------------------
// brute-force 1-NN in 3D with LDS-tiled reference points
// ------------------------------------------------------------------
__global__ void knn1_kernel(const float* __restrict__ ref, const float* __restrict__ q,
                            int* __restrict__ out, int nq, int nref) {
    __shared__ float tile[256 * 3];
    int qi = blockIdx.x * blockDim.x + threadIdx.x;
    float qx = 0.f, qy = 0.f, qz = 0.f;
    if (qi < nq) { qx = q[qi * 3 + 0]; qy = q[qi * 3 + 1]; qz = q[qi * 3 + 2]; }
    float best = 3.4e38f;
    int bidx = 0;
    for (int base = 0; base < nref; base += 256) {
        if (base + 256 < nref)
            __builtin_prefetch(ref + (size_t)(base + 256) * 3, 0, 1);
        for (int t = threadIdx.x; t < 256 * 3; t += blockDim.x)
            tile[t] = ref[base * 3 + t];
        __syncthreads();
#pragma unroll 4
        for (int j = 0; j < 256; ++j) {
            float dx = qx - tile[j * 3 + 0];
            float dy = qy - tile[j * 3 + 1];
            float dz = qz - tile[j * 3 + 2];
            float d = dx * dx + dy * dy + dz * dz;
            if (d < best) { best = d; bidx = base + j; }
        }
        __syncthreads();
    }
    if (qi < nq) out[qi] = bidx;
}

// ------------------------------------------------------------------
// logits = X[M x 64] @ W[22 x 64]^T + b via WMMA, N padded to 32.
// Block: 256 thr = 8 waves, covers 64 rows (4 M-tiles) x 32 cols (2 N-tiles).
// Weight tile staged zero-padded in LDS once -> no predication in loop.
// ------------------------------------------------------------------
__global__ void logits_wmma(const float* __restrict__ X, const float* __restrict__ Wt,
                            const float* __restrict__ bias, float* __restrict__ OUT) {
    __shared__ unsigned int ldsA[64 * 32];   // 64 rows x 64 K halfs (32 dwords/row)
    __shared__ unsigned int ldsB[32 * 32];   // 32 cols x 64 K halfs
    const int tid   = threadIdx.x;
    const int lane  = tid & 31;
    const int wv    = tid >> 5;         // 0..7
    const int mSub  = wv >> 1;          // 0..3
    const int nTile = wv & 1;           // 0..1
    const int mloc  = lane & 15;
    const int hi    = lane >> 4;
    const int mBase = blockIdx.x * 64;

    // zero-pad B tile, then fill 22 real columns (coalesced float4)
    for (int t = tid; t < 32 * 32; t += 256) ldsB[t] = 0u;
    __syncthreads();
    if (tid < 22 * 16) {
        int cidx = tid >> 4;            // 0..21
        int kq   = (tid & 15) * 4;      // 0..60
        float4 v = *(const float4*)(Wt + (size_t)cidx * CB + kq);
        unsigned int* dst = &ldsB[cidx * 32 + (kq >> 1)];
        dst[0] = pack2h(v.x, v.y);
        dst[1] = pack2h(v.z, v.w);
    }
    // fill A tile: 64 rows x 64 K f32 = 1024 float4
    for (int t = tid; t < 1024; t += 256) {
        int r  = t >> 4;
        int kq = (t & 15) * 4;
        float4 v = *(const float4*)(X + (size_t)(mBase + r) * CB + kq);
        unsigned int* dst = &ldsA[r * 32 + (kq >> 1)];
        dst[0] = pack2h(v.x, v.y);
        dst[1] = pack2h(v.z, v.w);
    }
    __syncthreads();

    v8f c = {};
#pragma unroll
    for (int kk = 0; kk < CB; kk += 32) {
        v16h a = frag_from_lds(&ldsA[(mSub * 16 + mloc) * 32 + (kk >> 1) + hi * 4]);
        v16h b = frag_from_lds(&ldsB[(nTile * 16 + mloc) * 32 + (kk >> 1) + hi * 4]);
        c = __builtin_amdgcn_wmma_f32_16x16x32_f16(false, a, false, b,
                                                   (short)0, c, false, false);
    }
    const int n = nTile * 16 + mloc;
    if (n < NCLS) {
        float bv = bias[n];
#pragma unroll
        for (int i = 0; i < 8; ++i) {
            int row = mBase + mSub * 16 + hi * 8 + i;
            OUT[row * NCLS + n] = c[i] + bv;
        }
    }
}

// ------------------------------------------------------------------
// Y[M x N] = X[M x K] @ W[N x K]^T + bias with fused BN column stats.
// Block: 256 thr = 8 waves covering 16 rows x 128 cols; A tile shared
// by all 8 waves per K-step (8x operand reuse from LDS).
// M mult of 16, N mult of 128, K mult of 32.
// ------------------------------------------------------------------
__global__ void gemm_wmma_bias_stats(const float* __restrict__ X,
                                     const float* __restrict__ W,
                                     const float* __restrict__ bias,
                                     float* __restrict__ Y,
                                     float* __restrict__ colsum,
                                     float* __restrict__ colsq,
                                     int M, int N, int K) {
    __shared__ unsigned int ldsA[16 * 16];    // 16 rows x 32 K halfs
    __shared__ unsigned int ldsB[128 * 16];   // 128 cols x 32 K halfs
    const int tid    = threadIdx.x;
    const int lane   = tid & 31;
    const int wv     = tid >> 5;              // 0..7 -> N sub-tile
    const int mloc   = lane & 15;
    const int hi     = lane >> 4;
    const int nBlks  = N >> 7;                // N/128
    const int mTile  = blockIdx.x / nBlks;
    const int nBase  = (blockIdx.x % nBlks) * 128;

    v8f c = {};
    for (int kk = 0; kk < K; kk += 32) {
        __syncthreads();
        // A tile: 16 rows x 32 K f32 = 128 float4 (threads 0..127)
        if (tid < 128) {
            int r  = tid >> 3;
            int kq = (tid & 7) * 4;
            float4 v = *(const float4*)(X + (size_t)(mTile * 16 + r) * K + kk + kq);
            unsigned int* dst = &ldsA[r * 16 + (kq >> 1)];
            dst[0] = pack2h(v.x, v.y);
            dst[1] = pack2h(v.z, v.w);
        }
        // B tile: 128 cols x 32 K f32 = 1024 float4 (256 thr x 4)
        for (int t = tid; t < 1024; t += 256) {
            int cidx = t >> 3;
            int kq   = (t & 7) * 4;
            float4 v = *(const float4*)(W + (size_t)(nBase + cidx) * K + kk + kq);
            unsigned int* dst = &ldsB[cidx * 16 + (kq >> 1)];
            dst[0] = pack2h(v.x, v.y);
            dst[1] = pack2h(v.z, v.w);
        }
        __syncthreads();
        v16h a = frag_from_lds(&ldsA[mloc * 16 + hi * 4]);
        v16h b = frag_from_lds(&ldsB[(wv * 16 + mloc) * 16 + hi * 4]);
        c = __builtin_amdgcn_wmma_f32_16x16x32_f16(false, a, false, b,
                                                   (short)0, c, false, false);
    }
    const int colB = nBase + wv * 16 + mloc;
    const float bv = bias[colB];
    float s = 0.f, sq = 0.f;
#pragma unroll
    for (int i = 0; i < 8; ++i) {
        float v = c[i] + bv;
        int row = mTile * 16 + hi * 8 + i;
        Y[(size_t)row * N + colB] = v;
        s  += v;
        sq += v * v;
    }
    atomicAdd(&colsum[colB], s);
    atomicAdd(&colsq[colB],  sq);
}

// ------------------------------------------------------------------
// BatchNorm (training, biased stats) + ReLU, elementwise
// ------------------------------------------------------------------
__global__ void bn_relu_kernel(const float* __restrict__ Y,
                               const float* __restrict__ csum,
                               const float* __restrict__ csq,
                               const float* __restrict__ g,
                               const float* __restrict__ be,
                               float* __restrict__ OUT, int M, int N) {
    int i = blockIdx.x * blockDim.x + threadIdx.x;
    if (i >= M * N) return;
    int nn = i % N;
    float inv_m = 1.f / (float)M;
    float mean = csum[nn] * inv_m;
    float var  = csq[nn] * inv_m - mean * mean;
    float x = (Y[i] - mean) * rsqrtf(var + BN_EPS) * g[nn] + be[nn];
    OUT[i] = fmaxf(x, 0.f);
}

// ------------------------------------------------------------------
// per-row CE + temperature-KL, block-reduced into acc[0], acc[1]
// ------------------------------------------------------------------
__global__ void ce_kl_kernel(const float* __restrict__ slog, const float* __restrict__ tlog,
                             const int* __restrict__ idx, const int* __restrict__ seg,
                             float* __restrict__ acc, int n) {
    int i = blockIdx.x * blockDim.x + threadIdx.x;
    float ce = 0.f, kl = 0.f;
    if (i < n) {
        const float* s = slog + (size_t)i * NCLS;
        float mx = -3.4e38f;
        for (int j = 0; j < NCLS; ++j) mx = fmaxf(mx, s[j]);
        float se = 0.f;
        for (int j = 0; j < NCLS; ++j) se += __expf(s[j] - mx);
        float lse = mx + __logf(se);
        ce = -(s[seg[i]] - lse);

        float mx2 = -3.4e38f;
        for (int j = 0; j < NCLS; ++j) mx2 = fmaxf(mx2, s[j] * 0.5f);
        float se2 = 0.f;
        for (int j = 0; j < NCLS; ++j) se2 += __expf(s[j] * 0.5f - mx2);
        float lse2 = mx2 + __logf(se2);

        const float* t = tlog + (size_t)idx[i] * NCLS;
        float mt = -3.4e38f;
        for (int j = 0; j < NCLS; ++j) mt = fmaxf(mt, t[j] * 0.5f);
        float st = 0.f;
        for (int j = 0; j < NCLS; ++j) st += __expf(t[j] * 0.5f - mt);
        float lst = __logf(st);
        float inv = 1.f / st;
        for (int j = 0; j < NCLS; ++j) {
            float e = __expf(t[j] * 0.5f - mt);
            float p = e * inv;
            if (p > 0.f) {
                float logp = (t[j] * 0.5f - mt) - lst;
                float slp  = s[j] * 0.5f - lse2;
                kl += p * (logp - slp);
            }
        }
    }
    __shared__ float red0[256];
    __shared__ float red1[256];
    red0[threadIdx.x] = ce;
    red1[threadIdx.x] = kl;
    __syncthreads();
    for (int off = 128; off > 0; off >>= 1) {
        if ((int)threadIdx.x < off) {
            red0[threadIdx.x] += red0[threadIdx.x + off];
            red1[threadIdx.x] += red1[threadIdx.x + off];
        }
        __syncthreads();
    }
    if (threadIdx.x == 0) {
        atomicAdd(&acc[0], red0[0]);
        atomicAdd(&acc[1], red1[0]);
    }
}

// ------------------------------------------------------------------
// MSE against kNN-gathered teacher features -> acc[2]
// ------------------------------------------------------------------
__global__ void mse_kernel(const float* __restrict__ X, const float* __restrict__ T,
                           const int* __restrict__ idx, float* __restrict__ acc,
                           int M, int N) {
    int i = blockIdx.x * blockDim.x + threadIdx.x;
    float v = 0.f;
    if (i < M * N) {
        int r = i / N, c = i % N;
        float d = X[i] - T[(size_t)idx[r] * N + c];
        v = d * d;
    }
    __shared__ float red[256];
    red[threadIdx.x] = v;
    __syncthreads();
    for (int off = 128; off > 0; off >>= 1) {
        if ((int)threadIdx.x < off) red[threadIdx.x] += red[threadIdx.x + off];
        __syncthreads();
    }
    if (threadIdx.x == 0) atomicAdd(&acc[2], red[0]);
}

__global__ void finalize_kernel(const float* __restrict__ acc, float* __restrict__ out) {
    if (threadIdx.x == 0) {
        float seg_loss = acc[0] / (float)N_PTS;
        float kl_loss  = 0.2f * (acc[1] / (float)N_PTS * 4.0f);   // TEMP^2 = 4
        float mse      = acc[2] / ((float)N_ENC * (float)T_DIM);
        out[0] = seg_loss + 0.01f * mse + kl_loss;
    }
}

// ------------------------------------------------------------------
// host side
// ------------------------------------------------------------------
extern "C" void kernel_launch(void* const* d_in, const int* in_sizes, int n_in,
                              void* d_out, int out_size, void* d_ws, size_t ws_size,
                              hipStream_t stream) {
    const float* s_feat_enc4 = (const float*)d_in[0];   // [8192,256]
    const float* t_feat_enc4 = (const float*)d_in[1];   // [8192,512]
    const float* sc_enc4     = (const float*)d_in[2];   // [8192,3]
    const float* tc_enc4     = (const float*)d_in[3];   // [8192,3]
    const float* s_feat_out  = (const float*)d_in[4];   // [16384,64]
    const float* t_feat_out  = (const float*)d_in[5];   // [16384,64]
    const float* s_coord     = (const float*)d_in[6];   // [16384,3]
    const float* t_coord     = (const float*)d_in[7];   // [16384,3]
    const int*   segment     = (const int*)d_in[8];     // [16384]
    const float* pW[5], *pb[5], *g[5], *be[5];
    for (int l = 0; l < 5; ++l) {
        pW[l] = (const float*)d_in[9 + 4 * l];
        pb[l] = (const float*)d_in[10 + 4 * l];
        g[l]  = (const float*)d_in[11 + 4 * l];
        be[l] = (const float*)d_in[12 + 4 * l];
    }
    const float* seg_W  = (const float*)d_in[29];
    const float* seg_b  = (const float*)d_in[30];
    const float* tseg_W = (const float*)d_in[31];
    const float* tseg_b = (const float*)d_in[32];

    // ---- workspace layout (floats) ----
    float* ws    = (float*)d_ws;
    float* acc   = ws;                       // acc0=CE, acc1=KL, acc2=MSE
    float* stats = ws + 16;                  // 5 layers x (sum[512] | sq[512])
    const int STATS_N = 16 + 5 * 1024;
    float* seg_logits = ws + 8192;                       // 16384*22
    float* t_logits   = seg_logits + (size_t)N_PTS * NCLS;
    float* buf0       = t_logits + (size_t)N_PTS * NCLS; // 8192*512 ping
    float* buf1       = buf0 + (size_t)N_ENC * T_DIM;    // 8192*512 pong
    int*   idx_pts    = (int*)(buf1 + (size_t)N_ENC * T_DIM);
    int*   idx_enc    = idx_pts + N_PTS;

    // 0) zero accumulators + BN stats (every call: graph-replay safe)
    zero_kernel<<<(STATS_N + 255) / 256, 256, 0, stream>>>(ws, STATS_N);

    // 1) kNN matches
    knn1_kernel<<<N_PTS / 256, 256, 0, stream>>>(t_coord, s_coord, idx_pts, N_PTS, N_PTS);
    knn1_kernel<<<N_ENC / 256, 256, 0, stream>>>(tc_enc4, sc_enc4, idx_enc, N_ENC, N_ENC);

    // 2) student / teacher seg logits (WMMA): 64 rows per block
    logits_wmma<<<N_PTS / 64, 256, 0, stream>>>(s_feat_out, seg_W, seg_b, seg_logits);
    logits_wmma<<<N_PTS / 64, 256, 0, stream>>>(t_feat_out, tseg_W, tseg_b, t_logits);

    // 3) CE + KL losses
    ce_kl_kernel<<<N_PTS / 256, 256, 0, stream>>>(seg_logits, t_logits, idx_pts,
                                                  segment, acc, N_PTS);

    // 4) 5-layer MLP: WMMA GEMM + fused BN stats, then BN+ReLU
    const int dims_in[5]  = {S_DIM, 128, 128, 128, 128};
    const int dims_out[5] = {128, 128, 128, 128, T_DIM};
    const float* x = s_feat_enc4;
    float* bufs[2] = {buf0, buf1};
    for (int l = 0; l < 5; ++l) {
        int K = dims_in[l], N = dims_out[l];
        float* y = bufs[l & 1];
        float* csum = stats + l * 1024;
        float* csq  = csum + 512;
        int blocks = (N_ENC / 16) * (N / 128);
        gemm_wmma_bias_stats<<<blocks, 256, 0, stream>>>(
            x, pW[l], pb[l], y, csum, csq, N_ENC, N, K);
        int elems = N_ENC * N;
        bn_relu_kernel<<<(elems + 255) / 256, 256, 0, stream>>>(
            y, csum, csq, g[l], be[l], y, N_ENC, N);
        x = y;
    }

    // 5) MSE feature distillation
    {
        int elems = N_ENC * T_DIM;
        mse_kernel<<<(elems + 255) / 256, 256, 0, stream>>>(
            x, t_feat_enc4, idx_enc, acc, N_ENC, T_DIM);
    }

    // 6) combine
    finalize_kernel<<<1, 32, 0, stream>>>(acc, (float*)d_out);
}
